// Graph_Generator_14671608283430
// MI455X (gfx1250) — matrix operand order repaired and checked by reading, not compiled, not run on hardware
//
#include <hip/hip_runtime.h>
#include <hip/hip_bf16.h>
#include <math.h>

// ---------------------------------------------------------------------------
// Graph_Generator for MI455X (gfx1250, wave32, WMMA 16x16x32 f16).
// Dominant cost: all-pairs edge MLP (640K edges x (129->128->128)) ~= 84 GFLOP
// over ~2 MB of live data -> compute bound -> fuse fully, run on v_wmma.
// Round 2: 2 neighbors per k-iter (2 independent WMMA chains, half the
// barriers), branch-free leaky-ReLU (max/min/fma), batched fragment loads.
// ---------------------------------------------------------------------------

typedef _Float16 v16h __attribute__((ext_vector_type(16)));
typedef _Float16 v8h  __attribute__((ext_vector_type(8)));
typedef float    v8f  __attribute__((ext_vector_type(8)));

union Frag16 { v16h v; v8h h[2]; _Float16 e[16]; };
union FragC  { v8f  v; float f[8]; };

#define WMMA_F16(A, B, C) \
  __builtin_amdgcn_wmma_f32_16x16x32_f16(false, (A), false, (B), (short)0, (C), false, false)

__device__ __forceinline__ float leaky02(float r) {
  return fmaxf(r, 0.0f) + 0.2f * fminf(r, 0.0f);   // v_max + v_min + v_fmac
}

// Problem constants
#define BB   64
#define NN   100
#define HH   64
#define MM   128
#define NJ   7        // ceil(100/16) j-tiles per batch

// ---------------------------------------------------------------------------
// Fused edge MLP + neighbor aggregation.
// Block = (batch b, 16-row j-tile). 8 waves; wave w owns output cols [16w,16w+16)
// of fe1 AND fe2, with its weight B-fragments held in VGPRs across the k-loop.
// dist term + bias folded into the WMMA accumulator init -> K is exactly 128.
// Two neighbors per iteration -> two independent accumulation chains.
// Writes xin[row][0:128]=agg, xin[row][128:192]=x  (GRU0 input, stride 192).
// ---------------------------------------------------------------------------
__global__ void __launch_bounds__(256)
edge_mlp_kernel(const float* __restrict__ x,
                const _Float16* __restrict__ W1,   // [128][128] f16 (fe1_w cols 0..127)
                const _Float16* __restrict__ W2,   // [128][128] f16
                const float* __restrict__ w1last,  // fe1_w[:,128]
                const float* __restrict__ b1,
                const float* __restrict__ b2,
                float* __restrict__ xin)
{
  __shared__ alignas(16) _Float16 xjA[16 * 64];     // A-matrix cols 0..63 (x_j), k-invariant
  __shared__ alignas(16) _Float16 xk[2 * 64];       // A-matrix cols 64..127 for 2 neighbors
  __shared__ alignas(16) _Float16 A2[2 * 16 * 128]; // fe1 activations (fe2 A-matrices)
  __shared__ float sdist[2 * 16];
  __shared__ float xj01[32];
  __shared__ float sb1[128], swl[128], sb2[128];

  const int tid = threadIdx.x;
  const int b  = blockIdx.x / NJ;
  const int j0 = (blockIdx.x % NJ) * 16;

  for (int e = tid; e < 16 * 64; e += 256) {
    int m = e >> 6, i = e & 63, j = j0 + m;
    xjA[e] = (j < NN) ? (_Float16)x[(b * NN + j) * HH + i] : (_Float16)0.0f;
  }
  if (tid < 128) { sb1[tid] = b1[tid]; swl[tid] = w1last[tid]; sb2[tid] = b2[tid]; }
  if (tid < 32) {
    int m = tid >> 1, c = tid & 1, j = j0 + m;
    xj01[tid] = (j < NN) ? x[(b * NN + j) * HH + c] : 0.0f;
  }
  __syncthreads();

  const int wave = tid >> 5, lane = tid & 31;
  const int hi = lane >> 4, lm = lane & 15;
  const int o = wave * 16 + lm;     // output column owned by this lane

  // Hoisted fragments: x_j A-frags (K chunks 0,32) + all weight B-frags.
  Frag16 axj[2], w1f[4], w2f[4];
#pragma unroll
  for (int c = 0; c < 2; ++c) {
    axj[c].h[0] = *(const v8h*)&xjA[lm * 64 + c * 32 + hi * 8];
    axj[c].h[1] = *(const v8h*)&xjA[lm * 64 + c * 32 + 16 + hi * 8];
  }
#pragma unroll
  for (int c = 0; c < 4; ++c) {
    w1f[c].h[0] = *(const v8h*)&W1[o * 128 + c * 32 + hi * 8];
    w1f[c].h[1] = *(const v8h*)&W1[o * 128 + c * 32 + 16 + hi * 8];
    w2f[c].h[0] = *(const v8h*)&W2[o * 128 + c * 32 + hi * 8];
    w2f[c].h[1] = *(const v8h*)&W2[o * 128 + c * 32 + 16 + hi * 8];
  }
  const float bias1 = sb1[o], wl = swl[o], bias2 = sb2[o];

  FragC agg;
#pragma unroll
  for (int v = 0; v < 8; ++v) agg.f[v] = 0.0f;

  for (int k = 0; k < NN; k += 2) {
    // Stage x_k for two neighbors + their distance columns.
    if (tid < 128) {
      int kk = tid >> 6, i = tid & 63;
      xk[kk * 64 + i] = (_Float16)x[(b * NN + k + kk) * HH + i];
    }
    if (tid < 32) {
      int kk = tid >> 4, m = tid & 15;
      float dx = x[(b * NN + k + kk) * HH + 0] - xj01[m * 2 + 0] + 1e-12f;
      float dy = x[(b * NN + k + kk) * HH + 1] - xj01[m * 2 + 1] + 1e-12f;
      sdist[kk * 16 + m] = sqrtf(dx * dx + dy * dy);
    }
    __syncthreads();

    // fe1 for both neighbors: two independent WMMA chains.
    FragC c1[2];
#pragma unroll
    for (int u = 0; u < 2; ++u) {
      Frag16 axk[2];   // x_k fragments: all 16 rows identical -> LDS broadcast
#pragma unroll
      for (int c = 0; c < 2; ++c) {
        axk[c].h[0] = *(const v8h*)&xk[u * 64 + c * 32 + hi * 8];
        axk[c].h[1] = *(const v8h*)&xk[u * 64 + c * 32 + 16 + hi * 8];
      }
#pragma unroll
      for (int v = 0; v < 8; ++v)
        c1[u].f[v] = bias1 + sdist[u * 16 + v + hi * 8] * wl;
      c1[u].v = WMMA_F16(axj[0].v, w1f[0].v, c1[u].v);
      c1[u].v = WMMA_F16(axj[1].v, w1f[1].v, c1[u].v);
      c1[u].v = WMMA_F16(axk[0].v, w1f[2].v, c1[u].v);
      c1[u].v = WMMA_F16(axk[1].v, w1f[3].v, c1[u].v);
    }
#pragma unroll
    for (int u = 0; u < 2; ++u)
#pragma unroll
      for (int v = 0; v < 8; ++v)
        A2[u * 2048 + (v + hi * 8) * 128 + o] = (_Float16)leaky02(c1[u].f[v]);
    __syncthreads();

    // fe2 for both neighbors; preload all A-frags, then chain WMMAs.
#pragma unroll
    for (int u = 0; u < 2; ++u) {
      Frag16 a2f[4];
#pragma unroll
      for (int c = 0; c < 4; ++c) {
        a2f[c].h[0] = *(const v8h*)&A2[u * 2048 + lm * 128 + c * 32 + hi * 8];
        a2f[c].h[1] = *(const v8h*)&A2[u * 2048 + lm * 128 + c * 32 + 16 + hi * 8];
      }
      FragC c2;
#pragma unroll
      for (int v = 0; v < 8; ++v) c2.f[v] = bias2;
      c2.v = WMMA_F16(a2f[0].v, w2f[0].v, c2.v);
      c2.v = WMMA_F16(a2f[1].v, w2f[1].v, c2.v);
      c2.v = WMMA_F16(a2f[2].v, w2f[2].v, c2.v);
      c2.v = WMMA_F16(a2f[3].v, w2f[3].v, c2.v);
#pragma unroll
      for (int v = 0; v < 8; ++v) agg.f[v] += leaky02(c2.f[v]);
    }
    // Next-iter xk/dist writes are fenced by this iter's mid barrier;
    // next-iter A2 writes are fenced by next iter's first barrier.
  }

  // xin[:,0:128] = agg
#pragma unroll
  for (int v = 0; v < 8; ++v) {
    int m = v + hi * 8, j = j0 + m;
    if (j < NN) xin[(b * NN + j) * 192 + o] = agg.f[v];
  }
  // xin[:,128:192] = x
  for (int e = tid; e < 16 * 64; e += 256) {
    int m = e >> 6, i = e & 63, j = j0 + m;
    if (j < NN) xin[(b * NN + j) * 192 + 128 + i] = x[(b * NN + j) * HH + i];
  }
}

// ---------------------------------------------------------------------------
// Generic WMMA GEMM: out[M,N] = act(A[M,K] @ W[N,K]^T + bias), A f32 -> f16 LDS.
// Block = 16-row M-tile, wave w covers N-tiles w, w+8, ... mode 1 = tanh.
// ---------------------------------------------------------------------------
__global__ void __launch_bounds__(256)
gemm_wmma_kernel(const float* __restrict__ A, int lda, int Mrows,
                 const _Float16* __restrict__ W, int K, int N,
                 const float* __restrict__ bias,
                 float* __restrict__ out, int ldo, int mode)
{
  __shared__ alignas(16) _Float16 As[16 * 192];
  const int tid = threadIdx.x;
  const int m0 = blockIdx.x * 16;

  const int total = 16 * K;
  for (int e = tid; e < total; e += 256) {
    int m = e / K, kk = e - m * K, gm = m0 + m;
    As[e] = (gm < Mrows) ? (_Float16)A[gm * lda + kk] : (_Float16)0.0f;
  }
  __syncthreads();

  const int wave = tid >> 5, lane = tid & 31;
  const int hi = lane >> 4, lm = lane & 15;
  const int nk = K >> 5;              // K chunks of 32 (K in {128,192})

  Frag16 aF[6];
  for (int c = 0; c < nk; ++c) {
    aF[c].h[0] = *(const v8h*)&As[lm * K + c * 32 + hi * 8];
    aF[c].h[1] = *(const v8h*)&As[lm * K + c * 32 + 16 + hi * 8];
  }

  const int ntiles = N >> 4;
  for (int nt = wave; nt < ntiles; nt += 8) {
    int o = nt * 16 + lm;
    Frag16 bF[6];                     // batch all B-frag global loads first
    for (int c = 0; c < nk; ++c) {
      bF[c].h[0] = *(const v8h*)&W[o * K + c * 32 + hi * 8];
      bF[c].h[1] = *(const v8h*)&W[o * K + c * 32 + 16 + hi * 8];
    }
    FragC acc;
    float bv = bias[o];
#pragma unroll
    for (int v = 0; v < 8; ++v) acc.f[v] = bv;
    for (int c = 0; c < nk; ++c)
      acc.v = WMMA_F16(aF[c].v, bF[c].v, acc.v);
#pragma unroll
    for (int v = 0; v < 8; ++v) {
      int gm = m0 + v + hi * 8;
      if (gm < Mrows) {
        float r = acc.f[v];
        if (mode == 1) r = tanhf(r);
        out[gm * ldo + o] = r;
      }
    }
  }
}

// ---------------------------------------------------------------------------
// Elementwise helpers
// ---------------------------------------------------------------------------
__global__ void gru_combine_kernel(const float* __restrict__ gx,
                                   const float* __restrict__ gh,
                                   float* __restrict__ h, int rows)
{
  int idx = blockIdx.x * blockDim.x + threadIdx.x;
  if (idx >= rows * MM) return;
  int m = idx >> 7, c = idx & 127;
  const float* gxr = gx + m * 3 * MM;
  const float* ghr = gh + m * 3 * MM;
  float r = 1.0f / (1.0f + __expf(-(gxr[c]        + ghr[c])));
  float z = 1.0f / (1.0f + __expf(-(gxr[MM + c]   + ghr[MM + c])));
  float n = tanhf(gxr[2 * MM + c] + r * ghr[2 * MM + c]);
  float hv = h[idx];
  h[idx] = n + z * (hv - n);
}

__global__ void cvt_f32_f16_kernel(const float* __restrict__ s, _Float16* __restrict__ d, int n)
{
  int i = blockIdx.x * blockDim.x + threadIdx.x;
  if (i < n) d[i] = (_Float16)s[i];
}

// fe1_w [128][129]: cols 0..127 -> f16 matrix, col 128 -> f32 w1last
__global__ void cvt_fe1_kernel(const float* __restrict__ s, _Float16* __restrict__ w,
                               float* __restrict__ wlast)
{
  int i = blockIdx.x * blockDim.x + threadIdx.x;
  if (i >= 128 * 129) return;
  int o = i / 129, c = i - o * 129;
  if (c < 128) w[o * 128 + c] = (_Float16)s[i];
  else         wlast[o] = s[i];
}

__global__ void zero_f32_kernel(float* __restrict__ p, int n)
{
  int i = blockIdx.x * blockDim.x + threadIdx.x;
  if (i < n) p[i] = 0.0f;
}

__global__ void final_out_kernel(const float* __restrict__ xb, float* __restrict__ out)
{
  int i = blockIdx.x * blockDim.x + threadIdx.x;
  if (i >= BB * NN * 3) return;
  int bn = i / 3, c = i - bn * 3;
  out[i] = xb[bn * HH + c];
}

// ---------------------------------------------------------------------------
// Workspace layout (bytes, 256-aligned blocks)
// ---------------------------------------------------------------------------
#define OFF_W1    ((size_t)0)                       // 128*128 f16
#define OFF_W2    (OFF_W1   + 32768)
#define OFF_X2H0  (OFF_W2   + 32768)                // 384*192 f16
#define OFF_H2H0  (OFF_X2H0 + 147456)               // 384*128 f16
#define OFF_X2H1  (OFF_H2H0 + 98304)
#define OFF_H2H1  (OFF_X2H1 + 98304)
#define OFF_FN2   (OFF_H2H1 + 98304)                // 64*128 f16
#define OFF_WL    (OFF_FN2  + 16384)                // 128 f32
#define OFF_XIN   (OFF_WL   + 512)                  // 6400*192 f32
#define OFF_GX    (OFF_XIN  + 4915200)              // 6400*384 f32
#define OFF_GH    (OFF_GX   + 9830400)
#define OFF_H0    (OFF_GH   + 9830400)              // 6400*128 f32
#define OFF_H1    (OFF_H0   + 3276800)
#define OFF_XB    (OFF_H1   + 3276800)              // 6400*64 f32

extern "C" void kernel_launch(void* const* d_in, const int* in_sizes, int n_in,
                              void* d_out, int out_size, void* d_ws, size_t ws_size,
                              hipStream_t stream)
{
  const float* x_in   = (const float*)d_in[0];
  const float* fe1_w  = (const float*)d_in[1];
  const float* fe1_b  = (const float*)d_in[2];
  const float* fe2_w  = (const float*)d_in[3];
  const float* fe2_b  = (const float*)d_in[4];
  const float* x2h_w0 = (const float*)d_in[5];
  const float* x2h_b0 = (const float*)d_in[6];
  const float* h2h_w0 = (const float*)d_in[7];
  const float* h2h_b0 = (const float*)d_in[8];
  const float* x2h_w1 = (const float*)d_in[9];
  const float* x2h_b1 = (const float*)d_in[10];
  const float* h2h_w1 = (const float*)d_in[11];
  const float* h2h_b1 = (const float*)d_in[12];
  const float* fn2_w  = (const float*)d_in[13];
  const float* fn2_b  = (const float*)d_in[14];

  char* ws = (char*)d_ws;
  _Float16* W1c   = (_Float16*)(ws + OFF_W1);
  _Float16* W2c   = (_Float16*)(ws + OFF_W2);
  _Float16* X2H0c = (_Float16*)(ws + OFF_X2H0);
  _Float16* H2H0c = (_Float16*)(ws + OFF_H2H0);
  _Float16* X2H1c = (_Float16*)(ws + OFF_X2H1);
  _Float16* H2H1c = (_Float16*)(ws + OFF_H2H1);
  _Float16* FN2c  = (_Float16*)(ws + OFF_FN2);
  float* w1last   = (float*)(ws + OFF_WL);
  float* xin      = (float*)(ws + OFF_XIN);
  float* gx       = (float*)(ws + OFF_GX);
  float* gh       = (float*)(ws + OFF_GH);
  float* h0       = (float*)(ws + OFF_H0);
  float* h1       = (float*)(ws + OFF_H1);
  float* xbuf     = (float*)(ws + OFF_XB);

  const int ROWS = BB * NN;   // 6400

  // Weight conversion f32 -> f16 (deterministic, cheap, graph-capture safe).
  cvt_fe1_kernel<<<(128 * 129 + 255) / 256, 256, 0, stream>>>(fe1_w, W1c, w1last);
  cvt_f32_f16_kernel<<<(16384 + 255) / 256, 256, 0, stream>>>(fe2_w, W2c, 16384);
  cvt_f32_f16_kernel<<<(73728 + 255) / 256, 256, 0, stream>>>(x2h_w0, X2H0c, 73728);
  cvt_f32_f16_kernel<<<(49152 + 255) / 256, 256, 0, stream>>>(h2h_w0, H2H0c, 49152);
  cvt_f32_f16_kernel<<<(49152 + 255) / 256, 256, 0, stream>>>(x2h_w1, X2H1c, 49152);
  cvt_f32_f16_kernel<<<(49152 + 255) / 256, 256, 0, stream>>>(h2h_w1, H2H1c, 49152);
  cvt_f32_f16_kernel<<<(8192 + 255) / 256, 256, 0, stream>>>(fn2_w, FN2c, 8192);
  zero_f32_kernel<<<(ROWS * MM + 255) / 256, 256, 0, stream>>>(h0, ROWS * MM);
  zero_f32_kernel<<<(ROWS * MM + 255) / 256, 256, 0, stream>>>(h1, ROWS * MM);

  for (int it = 0; it < 2; ++it) {
    const float* xsrc = (it == 0) ? x_in : xbuf;

    // Fused edge MLP + neighbor aggregation (the ~42 GFLOP/iter WMMA core).
    edge_mlp_kernel<<<BB * NJ, 256, 0, stream>>>(xsrc, W1c, W2c, w1last, fe1_b, fe2_b, xin);

    // GRU layer 0
    gemm_wmma_kernel<<<ROWS / 16, 256, 0, stream>>>(xin, 192, ROWS, X2H0c, 192, 384, x2h_b0, gx, 384, 0);
    gemm_wmma_kernel<<<ROWS / 16, 256, 0, stream>>>(h0, 128, ROWS, H2H0c, 128, 384, h2h_b0, gh, 384, 0);
    gru_combine_kernel<<<(ROWS * MM + 255) / 256, 256, 0, stream>>>(gx, gh, h0, ROWS);

    // GRU layer 1
    gemm_wmma_kernel<<<ROWS / 16, 256, 0, stream>>>(h0, 128, ROWS, X2H1c, 128, 384, x2h_b1, gx, 384, 0);
    gemm_wmma_kernel<<<ROWS / 16, 256, 0, stream>>>(h1, 128, ROWS, H2H1c, 128, 384, h2h_b1, gh, 384, 0);
    gru_combine_kernel<<<(ROWS * MM + 255) / 256, 256, 0, stream>>>(gx, gh, h1, ROWS);

    // x = tanh(h1 @ fn2_w^T + b)
    gemm_wmma_kernel<<<ROWS / 16, 256, 0, stream>>>(h1, 128, ROWS, FN2c, 128, 64, fn2_b, xbuf, 64, 1);
  }

  final_out_kernel<<<(BB * NN * 3 + 255) / 256, 256, 0, stream>>>(xbuf, (float*)d_out);
}
